// NTM_22462678958225
// MI455X (gfx1250) — matrix operand order repaired and checked by reading, not compiled, not run on hardware
//
#include <hip/hip_runtime.h>
#include <hip/hip_bf16.h>
#include <stdint.h>
#include <stddef.h>

// Problem dims
#define B_   128
#define N_   4096
#define M_   64
#define H_   512
#define NH_  4
#define NH2_ 8
#define IN_  256
#define OUT_ 256
#define GSZ  2048   // 4*H
#define XK   512    // IN + NH*M
#define CATK 768    // H + NH*M
#define EPS_ 1e-8f

typedef __attribute__((ext_vector_type(16))) __bf16 v16bf;
typedef __attribute__((ext_vector_type(8)))  float  v8f;

struct __attribute__((aligned(16))) U128 { unsigned int x, y, z, w; };
union FragBF { U128 q[2]; v16bf v; };

static __device__ __forceinline__ unsigned short f2bf(float f) {
  union { float f; unsigned int u; } cv; cv.f = f;
  unsigned int u = cv.u;
  unsigned int r = (u + 0x7FFFu + ((u >> 16) & 1u)) >> 16;
  return (unsigned short)r;
}
static __device__ __forceinline__ float sigm(float x) { return 1.f / (1.f + __expf(-x)); }
static __device__ __forceinline__ float softplus_(float x) {
  return (x > 20.f) ? x : log1pf(__expf(x));
}

// ---------------------------------------------------------------------------
// fp32 -> bf16 conversion (grid-stride)
// ---------------------------------------------------------------------------
__global__ void NTM_cvt_bf16(const float* __restrict__ s, unsigned short* __restrict__ d, int n) {
  for (int i = blockIdx.x * blockDim.x + threadIdx.x; i < n; i += gridDim.x * blockDim.x)
    d[i] = f2bf(s[i]);
}

// x = concat(in_data[B,256], swapaxes(prev_reads)[B, NH*M]) -> bf16 [B, 512]
__global__ void NTM_pack_x(const float* __restrict__ in_data,
                           const float* __restrict__ prev_reads,
                           unsigned short* __restrict__ x_bf) {
  int idx = blockIdx.x * blockDim.x + threadIdx.x;       // B * XK
  if (idx >= B_ * XK) return;
  int b = idx >> 9, col = idx & 511;
  float v;
  if (col < IN_) v = in_data[b * IN_ + col];
  else {
    int k = col - IN_; int i = k >> 6; int m = k & 63;
    v = prev_reads[((size_t)i * B_ + b) * M_ + m];
  }
  x_bf[idx] = f2bf(v);
}

// cat = concat(h[B,512], reads[4,B,64]) -> bf16 [B, 768]
__global__ void NTM_pack_cat(const float* __restrict__ h,
                             const float* __restrict__ reads,
                             unsigned short* __restrict__ cat_bf) {
  int idx = blockIdx.x * blockDim.x + threadIdx.x;       // B * CATK
  if (idx >= B_ * CATK) return;
  int b = idx / CATK, col = idx % CATK;
  float v;
  if (col < H_) v = h[b * H_ + col];
  else {
    int k = col - H_; int slot = k >> 6; int m = k & 63;
    v = reads[((size_t)slot * B_ + b) * M_ + m];
  }
  cat_bf[idx] = f2bf(v);
}

// ---------------------------------------------------------------------------
// Generic bf16 WMMA GEMM:  C[M,N] = act( A1@B1^T + A2@B2^T + bias1 + bias2 )
// A row-major [M,K], B row-major [N,K] (i.e. weight[out,in]).
// One wave32 per 16x16 tile; K-loop of v_wmma_f32_16x16x32_bf16.
// act: 0 none, 1 sigmoid, 2 tanh
// ---------------------------------------------------------------------------
__global__ void NTM_wmma_gemm(const unsigned short* __restrict__ A1,
                              const unsigned short* __restrict__ B1, int K1,
                              const unsigned short* __restrict__ A2,
                              const unsigned short* __restrict__ B2, int K2,
                              const float* __restrict__ bias1,
                              const float* __restrict__ bias2,
                              float* __restrict__ C, int Mtot, int Ntot, int act) {
  const int lane = threadIdx.x & 31;
  const int wid  = blockIdx.x * (blockDim.x >> 5) + (threadIdx.x >> 5);
  const int tilesN = Ntot >> 4;
  const int tiles  = (Mtot >> 4) * tilesN;
  if (wid >= tiles) return;                 // wave-uniform: EXEC stays all-ones
  const int m0 = (wid / tilesN) << 4;
  const int n0 = (wid % tilesN) << 4;

  const int rowA = m0 + (lane & 15);
  const int rowB = n0 + (lane & 15);
  const int hiA  = (lane >> 4) << 3;        // A 16-bit layout: K chunks at +hiA, +hiA+16
  const int hiB  = (lane >> 4) << 3;        // B 16-bit layout: K chunks at +hiB, +hiB+16

  v8f acc = {0.f, 0.f, 0.f, 0.f, 0.f, 0.f, 0.f, 0.f};

  for (int k0 = 0; k0 < K1; k0 += 32) {
    FragBF fa, fb;
    const unsigned short* pa = A1 + (size_t)rowA * K1 + k0 + hiA;
    fa.q[0] = *(const U128*)pa;
    fa.q[1] = *(const U128*)(pa + 16);
    const unsigned short* pb = B1 + (size_t)rowB * K1 + k0 + hiB;
    fb.q[0] = *(const U128*)pb;
    fb.q[1] = *(const U128*)(pb + 16);
    acc = __builtin_amdgcn_wmma_f32_16x16x32_bf16(false, fa.v, false, fb.v,
                                                  (short)0, acc, false, false);
  }
  if (A2 != nullptr) {
    for (int k0 = 0; k0 < K2; k0 += 32) {
      FragBF fa, fb;
      const unsigned short* pa = A2 + (size_t)rowA * K2 + k0 + hiA;
      fa.q[0] = *(const U128*)pa;
      fa.q[1] = *(const U128*)(pa + 16);
      const unsigned short* pb = B2 + (size_t)rowB * K2 + k0 + hiB;
      fb.q[0] = *(const U128*)pb;
      fb.q[1] = *(const U128*)(pb + 16);
      acc = __builtin_amdgcn_wmma_f32_16x16x32_bf16(false, fa.v, false, fb.v,
                                                    (short)0, acc, false, false);
    }
  }

  const int col   = n0 + (lane & 15);
  const int rbase = m0 + ((lane >> 4) << 3);  // +8 for high half-wave
  float badd = 0.f;
  if (bias1) badd += bias1[col];
  if (bias2) badd += bias2[col];
#pragma unroll
  for (int r = 0; r < 8; ++r) {
    float v = acc[r] + badd;
    if (act == 1) v = sigm(v);
    else if (act == 2) v = tanhf(v);
    C[(size_t)(rbase + r) * Ntot + col] = v;
  }
}

// ---------------------------------------------------------------------------
// LSTM elementwise: c = sig(gf)*c_prev + sig(gi)*tanh(gg); h = sig(go)*tanh(c)
// ---------------------------------------------------------------------------
__global__ void NTM_lstm_elem(const float* __restrict__ gates,
                              const float* __restrict__ c_prev,
                              float* __restrict__ c_out, float* __restrict__ h_out,
                              unsigned short* __restrict__ c_bf) {
  int idx = blockIdx.x * blockDim.x + threadIdx.x;    // B*H
  if (idx >= B_ * H_) return;
  int b = idx >> 9, j = idx & 511;
  const float* gr = gates + (size_t)b * GSZ;
  float gi = gr[j], gf = gr[H_ + j], gg = gr[2 * H_ + j], go = gr[3 * H_ + j];
  float c = sigm(gf) * c_prev[idx] + sigm(gi) * tanhf(gg);
  float h = sigm(go) * tanhf(c);
  c_out[idx] = c; h_out[idx] = h; c_bf[idx] = f2bf(c);
}

// ---------------------------------------------------------------------------
// Per-head scalar params: beta, g, gamma, s(3-softmax). Block per (head,batch).
// ---------------------------------------------------------------------------
static __device__ __forceinline__ float reduce128(float* red, int t, float v) {
  red[t] = v; __syncthreads();
  for (int s = 64; s > 0; s >>= 1) { if (t < s) red[t] += red[t + s]; __syncthreads(); }
  float r = red[0]; __syncthreads();
  return r;
}

__global__ void NTM_head_scalars(const float* __restrict__ c,
                                 const float* __restrict__ Wbeta, const float* __restrict__ bbeta,
                                 const float* __restrict__ Wg,    const float* __restrict__ bg,
                                 const float* __restrict__ Ws,    const float* __restrict__ bs,
                                 const float* __restrict__ Wgam,  const float* __restrict__ bgam,
                                 float* __restrict__ beta_o, float* __restrict__ g_o,
                                 float* __restrict__ gam_o,  float* __restrict__ s_o) {
  __shared__ float red[128];
  int i = blockIdx.x >> 7, b = blockIdx.x & 127, t = threadIdx.x;
  const float* cb = c + (size_t)b * H_;
  float pb = 0, pg = 0, pgam = 0, ps0 = 0, ps1 = 0, ps2 = 0;
  for (int hh = t; hh < H_; hh += 128) {
    float cv = cb[hh];
    pb   += cv * Wbeta[i * H_ + hh];
    pg   += cv * Wg[i * H_ + hh];
    pgam += cv * Wgam[i * H_ + hh];
    ps0  += cv * Ws[(i * 3 + 0) * H_ + hh];
    ps1  += cv * Ws[(i * 3 + 1) * H_ + hh];
    ps2  += cv * Ws[(i * 3 + 2) * H_ + hh];
  }
  float db   = reduce128(red, t, pb);
  float dg   = reduce128(red, t, pg);
  float dgam = reduce128(red, t, pgam);
  float d0   = reduce128(red, t, ps0);
  float d1   = reduce128(red, t, ps1);
  float d2   = reduce128(red, t, ps2);
  if (t == 0) {
    int ib = i * B_ + b;
    beta_o[ib] = softplus_(db + bbeta[i]);
    g_o[ib]    = sigm(dg + bg[i]);
    gam_o[ib]  = 1.f + softplus_(dgam + bgam[i]);
    float e0 = d0 + bs[i * 3 + 0], e1 = d1 + bs[i * 3 + 1], e2 = d2 + bs[i * 3 + 2];
    float mx = fmaxf(e0, fmaxf(e1, e2));
    float x0 = __expf(e0 - mx), x1 = __expf(e1 - mx), x2 = __expf(e2 - mx);
    float inv = 1.f / (x0 + x1 + x2);
    s_o[ib * 3 + 0] = x0 * inv; s_o[ib * 3 + 1] = x1 * inv; s_o[ib * 3 + 2] = x2 * inv;
  }
}

// ---------------------------------------------------------------------------
// Addressing: one workgroup (256 thr) per batch element. Content similarity
// (coalesced: 16 lanes cooperate per memory row, shuffle-reduced), softmax,
// interpolate, circular shift, sharpen, normalize -> w; fused weighted read
// for read heads. Full 4096-wide weight vectors live in LDS.
// ---------------------------------------------------------------------------
__global__ void NTM_addressing(int head, int is_read, int read_slot,
                               const float* __restrict__ mem,
                               const float* __restrict__ keyv_all,   // [B, 512]
                               const float* __restrict__ beta_a,
                               const float* __restrict__ g_a,
                               const float* __restrict__ gam_a,
                               const float* __restrict__ s_a,
                               const float* __restrict__ prev_w,     // [NH2, B, N]
                               float* __restrict__ w_out,            // [B, N]
                               float* __restrict__ reads_out) {      // [NH, B, M]
  __shared__ float key[M_];
  __shared__ float red[256];
  __shared__ float zbuf[N_];
  __shared__ float wbuf[N_];
  int b = blockIdx.x, t = threadIdx.x;

  if (t < M_) key[t] = keyv_all[(size_t)b * 512 + head * M_ + t];
  __syncthreads();

  // key norm
  float kv = (t < M_) ? key[t] * key[t] : 0.f;
  red[t] = kv; __syncthreads();
  for (int s = 128; s > 0; s >>= 1) { if (t < s) red[t] += red[t + s]; __syncthreads(); }
  float knorm = sqrtf(red[0]); __syncthreads();

  int ib = head * B_ + b;
  float beta = beta_a[ib], gg = g_a[ib], gamma = gam_a[ib];
  float s0 = s_a[ib * 3 + 0], s1 = s_a[ib * 3 + 1], s2 = s_a[ib * 3 + 2];

  const float* memb = mem + (size_t)b * N_ * M_;

  // beta * cosine similarity: half-wave (16 lanes) per row -> coalesced
  // 256B row loads; dot & norm reduced across the 16-lane group via shfl_xor.
  {
    const int g = t >> 4;          // group 0..15 -> row n0+g
    const int q = t & 15;          // lane in group -> columns 4q..4q+3
    const float4 kreg = ((const float4*)key)[q];
    for (int n0 = 0; n0 < N_; n0 += 16) {
      int n = n0 + g;
      if (n0 + 80 < N_)
        __builtin_prefetch(memb + (size_t)(n + 80) * M_ + q * 4, 0, 0);
      float4 v = *(const float4*)(memb + (size_t)n * M_ + q * 4);
      float dot = v.x * kreg.x + v.y * kreg.y + v.z * kreg.z + v.w * kreg.w;
      float nr  = v.x * v.x + v.y * v.y + v.z * v.z + v.w * v.w;
#pragma unroll
      for (int off = 8; off > 0; off >>= 1) {
        dot += __shfl_xor(dot, off, 32);
        nr  += __shfl_xor(nr,  off, 32);
      }
      if (q == 0) zbuf[n] = beta * dot / (sqrtf(nr) * knorm + EPS_);
    }
  }
  __syncthreads();

  // softmax over N
  float lm = -1e30f;
  for (int n = t; n < N_; n += 256) lm = fmaxf(lm, zbuf[n]);
  red[t] = lm; __syncthreads();
  for (int s = 128; s > 0; s >>= 1) { if (t < s) red[t] = fmaxf(red[t], red[t + s]); __syncthreads(); }
  float zmax = red[0]; __syncthreads();

  float lsum = 0.f;
  for (int n = t; n < N_; n += 256) { float e = __expf(zbuf[n] - zmax); zbuf[n] = e; lsum += e; }
  red[t] = lsum; __syncthreads();
  for (int s = 128; s > 0; s >>= 1) { if (t < s) red[t] += red[t + s]; __syncthreads(); }
  float inv = 1.f / red[0]; __syncthreads();

  // interpolate with previous weights
  const float* pw = prev_w + ((size_t)head * B_ + b) * N_;
  for (int n = t; n < N_; n += 256)
    wbuf[n] = gg * (zbuf[n] * inv) + (1.f - gg) * pw[n];
  __syncthreads();

  // circular shift + sharpen
  float psum = 0.f;
  for (int n = t; n < N_; n += 256) {
    float wsft = s0 * wbuf[(n + 1) & (N_ - 1)] + s1 * wbuf[n] + s2 * wbuf[(n - 1) & (N_ - 1)];
    float wp = __powf(wsft, gamma);
    zbuf[n] = wp; psum += wp;
  }
  red[t] = psum; __syncthreads();
  for (int s = 128; s > 0; s >>= 1) { if (t < s) red[t] += red[t + s]; __syncthreads(); }
  float invp = 1.f / (red[0] + EPS_); __syncthreads();

  float* wo = w_out + (size_t)b * N_;
  for (int n = t; n < N_; n += 256) { float w = zbuf[n] * invp; zbuf[n] = w; wo[n] = w; }
  __syncthreads();

  // fused weighted read (even heads): reads[m] = sum_n w[n]*mem[n,m]  (coalesced in m)
  if (is_read) {
    int m = t & 63, grp = t >> 6;
    float acc = 0.f;
    for (int n = grp; n < N_; n += 4) acc += zbuf[n] * memb[(size_t)n * M_ + m];
    red[t] = acc; __syncthreads();
    if (t < 64) {
      float r = red[t] + red[t + 64] + red[t + 128] + red[t + 192];
      reads_out[((size_t)read_slot * B_ + b) * M_ + t] = r;
    }
  }
}

// ---------------------------------------------------------------------------
// Write-head memory update: mem = mem*(1 - w⊗e) + w⊗a  (float4 per thread,
// separate kernel to keep 8M-thread parallelism for this L2-bandwidth pass)
// ---------------------------------------------------------------------------
__global__ void NTM_write_update(int head, float* __restrict__ mem,
                                 const float* __restrict__ w,      // [B, N]
                                 const float* __restrict__ e_all,  // [B, 512]
                                 const float* __restrict__ a_all) {
  size_t idx = (size_t)blockIdx.x * blockDim.x + threadIdx.x;
  const size_t total = (size_t)B_ * N_ * (M_ / 4);
  if (idx >= total) return;
  int m4 = (int)(idx & 15);
  size_t bn = idx >> 4;
  int n = (int)(bn & (N_ - 1));
  int b = (int)(bn >> 12);
  float wv = w[(size_t)b * N_ + n];
  const float4 e4 = *(const float4*)(e_all + (size_t)b * 512 + head * M_ + m4 * 4);
  const float4 a4 = *(const float4*)(a_all + (size_t)b * 512 + head * M_ + m4 * 4);
  float4* mp = (float4*)(mem + (((size_t)b * N_ + n) * M_ + m4 * 4));
  float4 mv = *mp;
  mv.x = mv.x * (1.f - wv * e4.x) + wv * a4.x;
  mv.y = mv.y * (1.f - wv * e4.y) + wv * a4.y;
  mv.z = mv.z * (1.f - wv * e4.z) + wv * a4.z;
  mv.w = mv.w * (1.f - wv * e4.w) + wv * a4.w;
  *mp = mv;
}

// ---------------------------------------------------------------------------
extern "C" void kernel_launch(void* const* d_in, const int* in_sizes, int n_in,
                              void* d_out, int out_size, void* d_ws, size_t ws_size,
                              hipStream_t stream) {
  (void)in_sizes; (void)n_in; (void)out_size; (void)ws_size;
  const float* in_data   = (const float*)d_in[0];
  const float* memory    = (const float*)d_in[1];
  const float* h_prev    = (const float*)d_in[2];
  const float* c_prev    = (const float*)d_in[3];
  const float* prev_rd   = (const float*)d_in[4];
  const float* prev_w    = (const float*)d_in[5];
  const float* W_ih      = (const float*)d_in[6];
  const float* b_ih      = (const float*)d_in[7];
  const float* W_hh      = (const float*)d_in[8];
  const float* b_hh      = (const float*)d_in[9];
  const float* W_out     = (const float*)d_in[10];
  const float* b_out     = (const float*)d_in[11];
  const float* Wk        = (const float*)d_in[12];
  const float* bk        = (const float*)d_in[13];
  const float* Wbeta     = (const float*)d_in[14];
  const float* bbeta     = (const float*)d_in[15];
  const float* Wg        = (const float*)d_in[16];
  const float* bg        = (const float*)d_in[17];
  const float* Ws        = (const float*)d_in[18];
  const float* bs        = (const float*)d_in[19];
  const float* Wgam      = (const float*)d_in[20];
  const float* bgam      = (const float*)d_in[21];
  const float* We        = (const float*)d_in[22];
  const float* be        = (const float*)d_in[23];
  const float* Wa        = (const float*)d_in[24];
  const float* ba        = (const float*)d_in[25];
  float* out = (float*)d_out;

  char* ws = (char*)d_ws;
  size_t off = 0;
  auto give = [&](size_t bytes) -> char* {
    char* p = ws + off; off += (bytes + 255) & ~(size_t)255; return p;
  };
  float*          memw   = (float*)give((size_t)B_ * N_ * M_ * 4);      // 134 MB, L2-resident
  unsigned short* Wih_bf = (unsigned short*)give((size_t)GSZ * XK * 2);
  unsigned short* Whh_bf = (unsigned short*)give((size_t)GSZ * H_ * 2);
  unsigned short* Wk_bf  = (unsigned short*)give((size_t)512 * H_ * 2);
  unsigned short* We_bf  = (unsigned short*)give((size_t)512 * H_ * 2);
  unsigned short* Wa_bf  = (unsigned short*)give((size_t)512 * H_ * 2);
  unsigned short* Wo_bf  = (unsigned short*)give((size_t)OUT_ * CATK * 2);
  unsigned short* x_bf   = (unsigned short*)give((size_t)B_ * XK * 2);
  unsigned short* h_bf   = (unsigned short*)give((size_t)B_ * H_ * 2);
  unsigned short* c_bf   = (unsigned short*)give((size_t)B_ * H_ * 2);
  unsigned short* cat_bf = (unsigned short*)give((size_t)B_ * CATK * 2);
  float* gates = (float*)give((size_t)B_ * GSZ * 4);
  float* c_f   = (float*)give((size_t)B_ * H_ * 4);
  float* h_f   = (float*)give((size_t)B_ * H_ * 4);
  float* keyv  = (float*)give((size_t)B_ * 512 * 4);
  float* e_f   = (float*)give((size_t)B_ * 512 * 4);
  float* a_f   = (float*)give((size_t)B_ * 512 * 4);
  float* beta  = (float*)give((size_t)NH2_ * B_ * 4);
  float* gint  = (float*)give((size_t)NH2_ * B_ * 4);
  float* gam   = (float*)give((size_t)NH2_ * B_ * 4);
  float* svec  = (float*)give((size_t)NH2_ * B_ * 3 * 4);
  float* wcur  = (float*)give((size_t)B_ * N_ * 4);
  float* reads = (float*)give((size_t)NH_ * B_ * M_ * 4);

  // working copy of memory (inputs must not be mutated)
  hipMemcpyAsync(memw, memory, (size_t)B_ * N_ * M_ * 4, hipMemcpyDeviceToDevice, stream);

  // weight / activation conversions to bf16
  NTM_cvt_bf16<<<256, 256, 0, stream>>>(W_ih,  Wih_bf, GSZ * XK);
  NTM_cvt_bf16<<<256, 256, 0, stream>>>(W_hh,  Whh_bf, GSZ * H_);
  NTM_cvt_bf16<<<256, 256, 0, stream>>>(Wk,    Wk_bf,  512 * H_);
  NTM_cvt_bf16<<<256, 256, 0, stream>>>(We,    We_bf,  512 * H_);
  NTM_cvt_bf16<<<256, 256, 0, stream>>>(Wa,    Wa_bf,  512 * H_);
  NTM_cvt_bf16<<<256, 256, 0, stream>>>(W_out, Wo_bf,  OUT_ * CATK);
  NTM_cvt_bf16<<<128, 256, 0, stream>>>(h_prev, h_bf,  B_ * H_);
  NTM_pack_x<<<(B_ * XK + 255) / 256, 256, 0, stream>>>(in_data, prev_rd, x_bf);

  // controller gates: [128,2048] = x@W_ih^T + h@W_hh^T + b_ih + b_hh
  NTM_wmma_gemm<<<128, 256, 0, stream>>>(x_bf, Wih_bf, XK, h_bf, Whh_bf, H_,
                                         b_ih, b_hh, gates, B_, GSZ, 0);
  NTM_lstm_elem<<<(B_ * H_ + 255) / 256, 256, 0, stream>>>(gates, c_prev, c_f, h_f, c_bf);

  // head parameter GEMMs off c: keyv (none), erase (sigmoid), add (tanh)
  NTM_wmma_gemm<<<32, 256, 0, stream>>>(c_bf, Wk_bf, H_, nullptr, nullptr, 0,
                                        bk, nullptr, keyv, B_, 512, 0);
  NTM_wmma_gemm<<<32, 256, 0, stream>>>(c_bf, We_bf, H_, nullptr, nullptr, 0,
                                        be, nullptr, e_f, B_, 512, 1);
  NTM_wmma_gemm<<<32, 256, 0, stream>>>(c_bf, Wa_bf, H_, nullptr, nullptr, 0,
                                        ba, nullptr, a_f, B_, 512, 2);
  NTM_head_scalars<<<NH2_ * B_, 128, 0, stream>>>(c_f, Wbeta, bbeta, Wg, bg, Ws, bs,
                                                  Wgam, bgam, beta, gint, gam, svec);

  // sequential heads (write heads mutate memw seen by later heads)
  for (int i = 0; i < NH2_; ++i) {
    int is_read = (i % 2 == 0);
    NTM_addressing<<<B_, 256, 0, stream>>>(i, is_read, i / 2, memw, keyv,
                                           beta, gint, gam, svec, prev_w, wcur, reads);
    if (!is_read) {
      const size_t tot = (size_t)B_ * N_ * (M_ / 4);
      NTM_write_update<<<(unsigned)((tot + 255) / 256), 256, 0, stream>>>(i, memw, wcur, e_f, a_f);
    }
  }

  // output layer: sigmoid([h ++ reads] @ W_out^T + b_out)
  NTM_pack_cat<<<(B_ * CATK + 255) / 256, 256, 0, stream>>>(h_f, reads, cat_bf);
  NTM_wmma_gemm<<<16, 256, 0, stream>>>(cat_bf, Wo_bf, CATK, nullptr, nullptr, 0,
                                        b_out, nullptr, out, B_, OUT_, 1);
}
